// LSTM_Stock_45767171506355
// MI455X (gfx1250) — matrix operand order repaired and checked by reading, not compiled, not run on hardware
//
#include <hip/hip_runtime.h>

typedef __attribute__((ext_vector_type(16))) _Float16 v16h;
typedef __attribute__((ext_vector_type(8)))  float    v8f;
typedef __attribute__((__vector_size__(4 * sizeof(int)))) int i32x4_;

#define B_   256
#define T_   512
#define I_   64
#define H_   512
#define KC   576     // I_ + H_ (concat K dimension)
#define G4   2048    // 4*H_
#define KPH  288     // K per LDS staging phase (2 phases)
#define LDSROW (KPH + 8)   // +8 f16 (16B) pad: row stride 148 dwords -> conflict-free

#if defined(__has_builtin)
#  if __has_builtin(__builtin_amdgcn_global_load_async_to_lds_b128)
#    define HAVE_ASYNC_LDS 1
#  endif
#  if __has_builtin(__builtin_amdgcn_s_wait_asynccnt)
#    define HAVE_WAIT_ASYNC 1
#  endif
#endif
#ifndef HAVE_ASYNC_LDS
#  define HAVE_ASYNC_LDS 0
#endif
#ifndef HAVE_WAIT_ASYNC
#  define HAVE_WAIT_ASYNC 0
#endif

union Frag16 { uint4 q[2]; v16h v; };

// ---------- prep: pack [W_ih | W_hh] into f16 Wcat[2048][576] ----------
__global__ void prep_w(const float* __restrict__ Wih, const float* __restrict__ Whh,
                       _Float16* __restrict__ Wcat) {
    int idx = blockIdx.x * 256 + threadIdx.x;
    if (idx >= G4 * KC) return;
    int g = idx / KC;
    int k = idx - g * KC;
    float v = (k < I_) ? Wih[g * I_ + k] : Whh[g * H_ + (k - I_)];
    Wcat[idx] = (_Float16)v;
}

__global__ void prep_bias(const float* __restrict__ bih, const float* __restrict__ bhh,
                          float* __restrict__ bias) {
    int g = blockIdx.x * 256 + threadIdx.x;
    if (g < G4) bias[g] = bih[g] + bhh[g];
}

// ---------- init: A0 = [x_0 | 0], c = 0 ----------
__global__ void init_state(const float* __restrict__ x, _Float16* __restrict__ A0,
                           float* __restrict__ c) {
    int idx = blockIdx.x * 256 + threadIdx.x;
    if (idx < B_ * KC) {
        int b = idx / KC;
        int k = idx - b * KC;
        A0[idx] = (k < I_) ? (_Float16)x[(b * T_ + 0) * I_ + k] : (_Float16)0.0f;
    }
    if (idx < B_ * H_) c[idx] = 0.0f;
}

__device__ __forceinline__ float sigmoidf_(float v) {
    return 1.0f / (1.0f + __expf(-v));
}

// ---------- fused LSTM step: gates GEMM (WMMA, LDS-staged weights) ----------
// grid = 64 blocks x 256 threads. Block bb: n-tile = bb&31, m-tiles
// (bb>>5)*8 .. +7 (one per wave). All 8 waves share the block's four
// gate-weight tiles, staged (async DMA if available) through LDS in two K-phases.
__global__ void __launch_bounds__(256) lstm_step(
    const float*    __restrict__ x,
    const _Float16* __restrict__ Wcat,
    const float*    __restrict__ bias,
    const _Float16* __restrict__ Acur,
    _Float16*       __restrict__ Anext,
    float*          __restrict__ c,
    float*          __restrict__ hf32,
    int t)
{
    __shared__ alignas(16) _Float16 Bs[64 * LDSROW];   // 4 gates x 16 cols, 288 K

    const int tid   = threadIdx.x;
    const int lane  = tid & 31;
    const int w     = tid >> 5;
    const int nt    = blockIdx.x & 31;            // hidden tile 0..31
    const int mt    = (blockIdx.x >> 5) * 8 + w;  // batch tile 0..15
    const int m0    = mt * 16;
    const int n0    = nt * 16;
    const int nloc  = lane & 15;
    const int half  = lane >> 4;
    const int nglob = n0 + nloc;

    // A-matrix 16x32 f16 fragment (this code's K order, same as B): m = m0 + (lane&15); lane-half selects
    // contiguous K: VGPR0-7 hold K = half*16+0..15
    const _Float16* pa = Acur + (size_t)(m0 + nloc) * KC + half * 16;
    // B fragments come from LDS: row (g*16 + nloc), contiguous 16 K per lane-half
    const _Float16* pbs = Bs + nloc * LDSROW + half * 16;

    v8f acc0 = {}, acc1 = {}, acc2 = {}, acc3 = {};   // i, f, g, o gate tiles

    for (int phase = 0; phase < 2; ++phase) {
        // ---- stage 64 rows x 288 K of Wcat into LDS (36 uint4 per row) ----
        #pragma unroll
        for (int j = 0; j < 9; ++j) {
            int elem   = j * 256 + tid;       // 0..2303
            int row    = elem / 36;           // 0..63 : g = row>>4, col = row&15
            int within = elem - row * 36;     // uint4 within row
            int g      = row >> 4;
            int ncol   = row & 15;
            const uint4* src = (const uint4*)(Wcat +
                (size_t)(g * H_ + n0 + ncol) * KC + phase * KPH) + within;
            uint4* dst = (uint4*)Bs + row * (LDSROW / 8) + within;
#if HAVE_ASYNC_LDS
            // CDNA5 async DMA: global -> LDS directly, tracked by ASYNCcnt
            __builtin_amdgcn_global_load_async_to_lds_b128(
                (__attribute__((address_space(1))) i32x4_*)(void*)src,
                (__attribute__((address_space(3))) i32x4_*)(void*)dst, 0, 0);
#else
            *dst = *src;                      // fallback: VGPR round-trip
#endif
        }
#if HAVE_ASYNC_LDS
#  if HAVE_WAIT_ASYNC
        __builtin_amdgcn_s_wait_asynccnt(0);
#  else
        asm volatile("s_wait_asynccnt 0x0" ::: "memory");
#  endif
#endif
        __syncthreads();

        // ---- 9 WMMA K-steps on this phase ----
        #pragma unroll 3
        for (int kk = 0; kk < KPH / 32; ++kk) {
            const int kg = phase * KPH + kk * 32;   // global K
            const int kl = kk * 32;                 // K within LDS phase
            Frag16 a, b0, b1, b2, b3;
            a.q[0]  = *(const uint4*)(pa + kg);
            a.q[1]  = *(const uint4*)(pa + kg + 8);
            b0.q[0] = *(const uint4*)(pbs + 0 * 16 * LDSROW + kl);
            b0.q[1] = *(const uint4*)(pbs + 0 * 16 * LDSROW + kl + 8);
            b1.q[0] = *(const uint4*)(pbs + 1 * 16 * LDSROW + kl);
            b1.q[1] = *(const uint4*)(pbs + 1 * 16 * LDSROW + kl + 8);
            b2.q[0] = *(const uint4*)(pbs + 2 * 16 * LDSROW + kl);
            b2.q[1] = *(const uint4*)(pbs + 2 * 16 * LDSROW + kl + 8);
            b3.q[0] = *(const uint4*)(pbs + 3 * 16 * LDSROW + kl);
            b3.q[1] = *(const uint4*)(pbs + 3 * 16 * LDSROW + kl + 8);
            acc0 = __builtin_amdgcn_wmma_f32_16x16x32_f16(false, a.v, false, b0.v,
                                                          (short)0, acc0, false, false);
            acc1 = __builtin_amdgcn_wmma_f32_16x16x32_f16(false, a.v, false, b1.v,
                                                          (short)0, acc1, false, false);
            acc2 = __builtin_amdgcn_wmma_f32_16x16x32_f16(false, a.v, false, b2.v,
                                                          (short)0, acc2, false, false);
            acc3 = __builtin_amdgcn_wmma_f32_16x16x32_f16(false, a.v, false, b3.v,
                                                          (short)0, acc3, false, false);
        }
        __syncthreads();   // protect LDS before next phase restages
    }

    const float bi = bias[0 * H_ + nglob];
    const float bf = bias[1 * H_ + nglob];
    const float bg = bias[2 * H_ + nglob];
    const float bo = bias[3 * H_ + nglob];

    // C/D 16x16 f32 layout: lane holds column n = lane&15; VGPR r holds row
    // m = r + 8*(lane>>4). Cell state c is exclusively owned by this wave.
    #pragma unroll
    for (int r = 0; r < 8; ++r) {
        const int brow = m0 + r + 8 * half;
        const int idx  = brow * H_ + nglob;
        const float ig = sigmoidf_(acc0[r] + bi);
        const float fg = sigmoidf_(acc1[r] + bf);
        const float gg = tanhf(acc2[r] + bg);
        const float og = sigmoidf_(acc3[r] + bo);
        const float cn = fg * c[idx] + ig * gg;
        const float hn = og * tanhf(cn);
        c[idx] = cn;
        Anext[(size_t)brow * KC + I_ + nglob] = (_Float16)hn;
        if (t == T_ - 1) hf32[idx] = hn;
    }

    // stage x_{t+1} into the next A buffer (nt==0 blocks; one wave per m-tile)
    if (nt == 0 && (t + 1) < T_) {
        const int row  = m0 + (lane >> 1);
        const int col0 = (lane & 1) * 32;
        const float* xp = x + ((size_t)row * T_ + (t + 1)) * I_ + col0;
        _Float16* ap = Anext + (size_t)row * KC + col0;
        #pragma unroll
        for (int i = 0; i < 32; ++i) ap[i] = (_Float16)xp[i];
    }
}

// ---------- heads: out[k][b][o] = W_heads[k][o][:] . h_T[b][:] + b_heads ----------
__global__ void heads_kernel(const float* __restrict__ hT, const float* __restrict__ Wh,
                             const float* __restrict__ bh, float* __restrict__ out) {
    int b = blockIdx.x * blockDim.x + threadIdx.x;
    if (b >= B_) return;
    float acc[18];
    #pragma unroll
    for (int j = 0; j < 18; ++j) acc[j] = 0.0f;
    for (int h = 0; h < H_; ++h) {
        float hv = hT[b * H_ + h];
        #pragma unroll
        for (int j = 0; j < 18; ++j) acc[j] += Wh[j * H_ + h] * hv;
    }
    #pragma unroll
    for (int k = 0; k < 6; ++k)
        #pragma unroll
        for (int o = 0; o < 3; ++o)
            out[(k * B_ + b) * 3 + o] = acc[k * 3 + o] + bh[k * 3 + o];
}

extern "C" void kernel_launch(void* const* d_in, const int* in_sizes, int n_in,
                              void* d_out, int out_size, void* d_ws, size_t ws_size,
                              hipStream_t stream) {
    const float* x   = (const float*)d_in[0];   // [256,512,64]
    const float* Wih = (const float*)d_in[1];   // [2048,64]
    const float* Whh = (const float*)d_in[2];   // [2048,512]
    const float* bih = (const float*)d_in[3];   // [2048]
    const float* bhh = (const float*)d_in[4];   // [2048]
    const float* Whd = (const float*)d_in[5];   // [6,3,512]
    const float* bhd = (const float*)d_in[6];   // [6,3]
    float* out = (float*)d_out;                 // [6,256,3]

    char* ws = (char*)d_ws;
    _Float16* Wcat = (_Float16*)ws;  ws += (size_t)G4 * KC * sizeof(_Float16);
    float*    bias = (float*)ws;     ws += (size_t)G4 * sizeof(float);
    _Float16* A0   = (_Float16*)ws;  ws += (size_t)B_ * KC * sizeof(_Float16);
    _Float16* A1   = (_Float16*)ws;  ws += (size_t)B_ * KC * sizeof(_Float16);
    float*    cbuf = (float*)ws;     ws += (size_t)B_ * H_ * sizeof(float);
    float*    hf   = (float*)ws;     ws += (size_t)B_ * H_ * sizeof(float);

    prep_w    <<<(G4 * KC + 255) / 256, 256, 0, stream>>>(Wih, Whh, Wcat);
    prep_bias <<<(G4 + 255) / 256,      256, 0, stream>>>(bih, bhh, bias);
    init_state<<<(B_ * KC + 255) / 256, 256, 0, stream>>>(x, A0, cbuf);

    for (int t = 0; t < T_; ++t) {
        const _Float16* Ac = (t & 1) ? A1 : A0;
        _Float16*       An = (t & 1) ? A0 : A1;
        lstm_step<<<64, 256, 0, stream>>>(x, Wcat, bias, Ac, An, cbuf, hf, t);
    }

    heads_kernel<<<1, 256, 0, stream>>>(hf, Whd, bhd, out);
}
